// Model_BinClassifier_16406775071384
// MI455X (gfx1250) — compile-verified, matrix operand order checked by reading, not run on hardware
//
#include <hip/hip_runtime.h>
#include <hip/hip_bf16.h>
#include <math.h>

// ---------------- problem constants (match reference) ----------------
#define F_IN   128
#define H1DIM  64          // HEADS * C1
#define HEADS  8
#define C1     8
#define NCLS   2
#define NEG_SLOPE 0.2f

typedef __attribute__((ext_vector_type(2)))  float    v2f;
typedef __attribute__((ext_vector_type(8)))  float    v8f;
typedef __attribute__((ext_vector_type(16))) _Float16 v16h;

#if __has_builtin(__builtin_amdgcn_wmma_f32_16x16x4_f32)
#define HAVE_WMMA_F32 1
#else
#define HAVE_WMMA_F32 0
#endif

// ---- monotone float<->uint keys: native global_atomic_max_u32 == float max ----
__device__ __forceinline__ unsigned fenc(float f) {
  unsigned b = __float_as_uint(f);
  return (b & 0x80000000u) ? ~b : (b | 0x80000000u);
}
__device__ __forceinline__ float fdec(unsigned k) {
  unsigned b = (k & 0x80000000u) ? (k & 0x7fffffffu) : ~k;
  return __uint_as_float(b);
}
__device__ __forceinline__ float lrelu(float v) { return v > 0.f ? v : NEG_SLOPE * v; }

// =====================================================================
// Kernel 1: h1 = x @ W1  via WMMA.  One wave computes a 16x64 tile.
// A layout (f32 16x16x4): lanes 0-15 hold K=k,k+1 ; lanes 16-31 K=k+2,k+3
// C layout: VGPR v -> row v (lanes 0-15) / row 8+v (lanes 16-31), col = lane%16
// =====================================================================
__global__ void k_gat_gemm1(const float* __restrict__ x, const float* __restrict__ W,
                            float* __restrict__ h1, int nrows) {
  const int lane = threadIdx.x & 31;
  const int wave = threadIdx.x >> 5;
  const int r0 = (blockIdx.x * 4 + wave) * 16;
  if (r0 >= nrows) return;                 // whole wave exits together (EXEC stays all-1)
  const int g  = lane >> 4;                // half-wave id
  const int lr = lane & 15;                // A-row / B-col within tile
  v8f acc0 = {}, acc1 = {}, acc2 = {}, acc3 = {};
  const float* xr = x + (size_t)(r0 + lr) * F_IN;
#if HAVE_WMMA_F32
  for (int k = 0; k < F_IN; k += 4) {
    const int ka = k + g * 2;
    v2f a; a.x = xr[ka]; a.y = xr[ka + 1];
    const float* w0 = W + (size_t)ka * H1DIM;
    const float* w1 = w0 + H1DIM;
    v2f b0, b1, b2, b3;
    b0.x = w0[lr];      b0.y = w1[lr];
    b1.x = w0[16 + lr]; b1.y = w1[16 + lr];
    b2.x = w0[32 + lr]; b2.y = w1[32 + lr];
    b3.x = w0[48 + lr]; b3.y = w1[48 + lr];
    acc0 = __builtin_amdgcn_wmma_f32_16x16x4_f32(false, a, false, b0, (short)0, acc0, false, false);
    acc1 = __builtin_amdgcn_wmma_f32_16x16x4_f32(false, a, false, b1, (short)0, acc1, false, false);
    acc2 = __builtin_amdgcn_wmma_f32_16x16x4_f32(false, a, false, b2, (short)0, acc2, false, false);
    acc3 = __builtin_amdgcn_wmma_f32_16x16x4_f32(false, a, false, b3, (short)0, acc3, false, false);
  }
#else
  // fallback: fp16 A/B, fp32 accumulate (codegen-confirmed builtin)
  for (int k = 0; k < F_IN; k += 32) {
    v16h a, b0, b1, b2, b3;
    #pragma unroll
    for (int v = 0; v < 8; ++v) {
      const int kb = k + (v & 3) * 2 + (v >> 2) * 16 + g * 8;
      a[2*v]   = (_Float16)xr[kb];
      a[2*v+1] = (_Float16)xr[kb + 1];
      const float* w0 = W + (size_t)kb * H1DIM;
      const float* w1 = w0 + H1DIM;
      b0[2*v] = (_Float16)w0[lr];      b0[2*v+1] = (_Float16)w1[lr];
      b1[2*v] = (_Float16)w0[16+lr];   b1[2*v+1] = (_Float16)w1[16+lr];
      b2[2*v] = (_Float16)w0[32+lr];   b2[2*v+1] = (_Float16)w1[32+lr];
      b3[2*v] = (_Float16)w0[48+lr];   b3[2*v+1] = (_Float16)w1[48+lr];
    }
    acc0 = __builtin_amdgcn_wmma_f32_16x16x32_f16(false, a, false, b0, (short)0, acc0, false, false);
    acc1 = __builtin_amdgcn_wmma_f32_16x16x32_f16(false, a, false, b1, (short)0, acc1, false, false);
    acc2 = __builtin_amdgcn_wmma_f32_16x16x32_f16(false, a, false, b2, (short)0, acc2, false, false);
    acc3 = __builtin_amdgcn_wmma_f32_16x16x32_f16(false, a, false, b3, (short)0, acc3, false, false);
  }
#endif
  #pragma unroll
  for (int v = 0; v < 8; ++v) {
    float* o = h1 + (size_t)(r0 + g * 8 + v) * H1DIM;
    o[lr]      = acc0[v];
    o[16 + lr] = acc1[v];
    o[32 + lr] = acc2[v];
    o[48 + lr] = acc3[v];
  }
}

// =====================================================================
// Kernel 2: per (node, head) alpha projections + init of accumulators.
// =====================================================================
__global__ void k_alpha1(const float* __restrict__ h1,
                         const float* __restrict__ asrc, const float* __restrict__ adst,
                         float* __restrict__ a_s, float* __restrict__ a_d,
                         unsigned* __restrict__ m_enc, float* __restrict__ denom,
                         float* __restrict__ out1, int n) {
  int t = blockIdx.x * blockDim.x + threadIdx.x;
  if (t >= n * HEADS) return;
  int node = t >> 3, h = t & 7;
  const float* hp = h1 + (size_t)node * H1DIM + h * C1;
  float s = 0.f, d = 0.f;
  #pragma unroll
  for (int c = 0; c < C1; ++c) { float v = hp[c]; s += v * asrc[h*C1+c]; d += v * adst[h*C1+c]; }
  a_s[t] = s; a_d[t] = d;
  m_enc[t] = fenc(-3.3e38f);
  denom[t] = 0.f;
  float* op = out1 + (size_t)node * H1DIM + h * C1;
  #pragma unroll
  for (int c = 0; c < C1; ++c) op[c] = 0.f;
}

// =====================================================================
// Kernel 3: segment-max over dst (pass A). One thread per (edge, head).
// Edge list = [edge_index | self loops] handled virtually.
// =====================================================================
__global__ void k_edge_max1(const int* __restrict__ ei, long long E, long long ET,
                            const float* __restrict__ a_s, const float* __restrict__ a_d,
                            unsigned* __restrict__ m_enc) {
  long long t = (long long)blockIdx.x * blockDim.x + threadIdx.x;
  if (t >= ET * HEADS) return;
  long long e = t >> 3; int h = (int)(t & 7);
  int s, d;
  if (e < E) { s = ei[e]; d = ei[E + e]; } else { s = d = (int)(e - E); }
  float v = lrelu(a_s[s * HEADS + h] + a_d[d * HEADS + h]);
  atomicMax(&m_enc[d * HEADS + h], fenc(v));
}

// =====================================================================
// Kernel 4: pass B — unnormalized weighted message + denominator.
// out[dst] += exp(e - m[dst]) * h1[src]; denom[dst] += exp(e - m[dst])
// (division by denom at the end == reference softmax exactly)
// =====================================================================
__global__ void k_edge_acc1(const int* __restrict__ ei, long long E, long long ET,
                            const float* __restrict__ a_s, const float* __restrict__ a_d,
                            const unsigned* __restrict__ m_enc,
                            const float* __restrict__ h1,
                            float* __restrict__ denom, float* __restrict__ out1) {
  long long t = (long long)blockIdx.x * blockDim.x + threadIdx.x;
  if (t >= ET * HEADS) return;
  long long e = t >> 3; int h = (int)(t & 7);
  int s, d;
  if (e < E) { s = ei[e]; d = ei[E + e]; } else { s = d = (int)(e - E); }
  const float* hs = h1 + (size_t)s * H1DIM + h * C1;
  __builtin_prefetch(hs, 0, 1);                       // global_prefetch_b8
  float v = lrelu(a_s[s * HEADS + h] + a_d[d * HEADS + h]);
  float w = __expf(v - fdec(m_enc[d * HEADS + h]));
  atomicAdd(&denom[d * HEADS + h], w);
  float* od = out1 + (size_t)d * H1DIM + h * C1;
  #pragma unroll
  for (int c = 0; c < C1; ++c) atomicAdd(&od[c], w * hs[c]);
}

// =====================================================================
// Kernel 5: finalize conv1: divide, +bias, relu (in place).
// =====================================================================
__global__ void k_final1(float* __restrict__ out1, const float* __restrict__ denom,
                         const float* __restrict__ b1, int n) {
  int t = blockIdx.x * blockDim.x + threadIdx.x;
  if (t >= n * H1DIM) return;
  int node = t >> 6, col = t & 63, h = col >> 3;
  float v = out1[t] / (denom[node * HEADS + h] + 1e-16f) + b1[col];
  out1[t] = v > 0.f ? v : 0.f;
}

// =====================================================================
// Kernel 6: conv2 projection (64 -> 2) + alphas + accumulator init.
// =====================================================================
__global__ void k_proj2(const float* __restrict__ hin, const float* __restrict__ W2,
                        const float* __restrict__ asrc2, const float* __restrict__ adst2,
                        float* __restrict__ h2, float* __restrict__ a_s, float* __restrict__ a_d,
                        unsigned* __restrict__ m_enc, float* __restrict__ denom,
                        float* __restrict__ out2, int n) {
  int t = blockIdx.x * blockDim.x + threadIdx.x;
  if (t >= n) return;
  const float* hp = hin + (size_t)t * H1DIM;
  float o0 = 0.f, o1 = 0.f;
  #pragma unroll 8
  for (int k = 0; k < H1DIM; ++k) { float v = hp[k]; o0 += v * W2[k*NCLS]; o1 += v * W2[k*NCLS+1]; }
  h2[t*NCLS] = o0; h2[t*NCLS+1] = o1;
  a_s[t] = o0 * asrc2[0] + o1 * asrc2[1];
  a_d[t] = o0 * adst2[0] + o1 * adst2[1];
  m_enc[t] = fenc(-3.3e38f);
  denom[t] = 0.f;
  out2[t*NCLS] = 0.f; out2[t*NCLS+1] = 0.f;
}

__global__ void k_edge_max2(const int* __restrict__ ei, long long E, long long ET,
                            const float* __restrict__ a_s, const float* __restrict__ a_d,
                            unsigned* __restrict__ m_enc) {
  long long t = (long long)blockIdx.x * blockDim.x + threadIdx.x;
  if (t >= ET) return;
  int s, d;
  if (t < E) { s = ei[t]; d = ei[E + t]; } else { s = d = (int)(t - E); }
  atomicMax(&m_enc[d], fenc(lrelu(a_s[s] + a_d[d])));
}

__global__ void k_edge_acc2(const int* __restrict__ ei, long long E, long long ET,
                            const float* __restrict__ a_s, const float* __restrict__ a_d,
                            const unsigned* __restrict__ m_enc,
                            const float* __restrict__ h2,
                            float* __restrict__ denom, float* __restrict__ out2) {
  long long t = (long long)blockIdx.x * blockDim.x + threadIdx.x;
  if (t >= ET) return;
  int s, d;
  if (t < E) { s = ei[t]; d = ei[E + t]; } else { s = d = (int)(t - E); }
  float w = __expf(lrelu(a_s[s] + a_d[d]) - fdec(m_enc[d]));
  atomicAdd(&denom[d], w);
  atomicAdd(&out2[d*NCLS],     w * h2[s*NCLS]);
  atomicAdd(&out2[d*NCLS + 1], w * h2[s*NCLS + 1]);
}

// =====================================================================
// Kernel 9: zero pooled sums/counts.
// =====================================================================
__global__ void k_pool_init(float* __restrict__ pooled, float* __restrict__ cnt, int g) {
  int t = threadIdx.x;
  if (t < g * NCLS) pooled[t] = 0.f;
  if (t < g)        cnt[t] = 0.f;
}

// =====================================================================
// Kernel 10: finalize conv2 per node + global mean-pool accumulation.
// (heads=1, concat=False -> mean over heads is identity)
// =====================================================================
__global__ void k_pool(const float* __restrict__ out2, const float* __restrict__ denom,
                       const float* __restrict__ b2, const int* __restrict__ batch,
                       float* __restrict__ pooled, float* __restrict__ cnt, int n) {
  int t = blockIdx.x * blockDim.x + threadIdx.x;
  if (t >= n) return;
  float inv = 1.f / (denom[t] + 1e-16f);
  float o0 = out2[t*NCLS] * inv + b2[0];
  float o1 = out2[t*NCLS+1] * inv + b2[1];
  int g = batch[t];
  atomicAdd(&pooled[g*NCLS],     o0);
  atomicAdd(&pooled[g*NCLS + 1], o1);
  atomicAdd(&cnt[g], 1.f);
}

// =====================================================================
// Kernel 11: mean + log_softmax over [G, 2] -> d_out.
// =====================================================================
__global__ void k_logsm(const float* __restrict__ pooled, const float* __restrict__ cnt,
                        float* __restrict__ out, int g) {
  int t = blockIdx.x * blockDim.x + threadIdx.x;
  if (t >= g) return;
  float c = cnt[t] > 1.f ? cnt[t] : 1.f;
  float s0 = pooled[t*NCLS] / c;
  float s1 = pooled[t*NCLS+1] / c;
  float m = s0 > s1 ? s0 : s1;
  float l = m + logf(__expf(s0 - m) + __expf(s1 - m));
  out[t*NCLS]   = s0 - l;
  out[t*NCLS+1] = s1 - l;
}

// =====================================================================
static inline int cdiv(long long a, long long b) { return (int)((a + b - 1) / b); }

extern "C" void kernel_launch(void* const* d_in, const int* in_sizes, int n_in,
                              void* d_out, int out_size, void* d_ws, size_t ws_size,
                              hipStream_t stream) {
  const float* x      = (const float*)d_in[0];
  const int*   ei     = (const int*)  d_in[1];
  const int*   batch  = (const int*)  d_in[2];
  const float* W1     = (const float*)d_in[3];
  const float* asrc1  = (const float*)d_in[4];
  const float* adst1  = (const float*)d_in[5];
  const float* b1     = (const float*)d_in[6];
  const float* W2     = (const float*)d_in[7];
  const float* asrc2  = (const float*)d_in[8];
  const float* adst2  = (const float*)d_in[9];
  const float* b2     = (const float*)d_in[10];
  float* out = (float*)d_out;

  const int       N  = in_sizes[0] / F_IN;
  const long long E  = in_sizes[1] / 2;
  const long long ET = E + N;
  const int       G  = out_size / NCLS;

  // ---- carve workspace (all 4-byte elements) ----
  float* base = (float*)d_ws;
  size_t off = 0;
  auto take = [&](size_t nelem) { float* p = base + off; off += nelem; return p; };
  float*    h1    = take((size_t)N * H1DIM);
  float*    as1   = take((size_t)N * HEADS);
  float*    ad1   = take((size_t)N * HEADS);
  unsigned* m1    = (unsigned*)take((size_t)N * HEADS);
  float*    dn1   = take((size_t)N * HEADS);
  float*    o1    = take((size_t)N * H1DIM);
  float*    h2    = take((size_t)N * NCLS);
  float*    as2   = take((size_t)N);
  float*    ad2   = take((size_t)N);
  unsigned* m2    = (unsigned*)take((size_t)N);
  float*    dn2   = take((size_t)N);
  float*    o2    = take((size_t)N * NCLS);
  float*    pooled= take((size_t)G * NCLS);
  float*    cnt   = take((size_t)G);
  (void)ws_size; (void)n_in;

  // ---- layer 1 ----
  k_gat_gemm1<<<cdiv(cdiv(N, 16), 4), 128, 0, stream>>>(x, W1, h1, N);
  k_alpha1<<<cdiv((long long)N * HEADS, 256), 256, 0, stream>>>(
      h1, asrc1, adst1, as1, ad1, m1, dn1, o1, N);
  k_edge_max1<<<cdiv(ET * HEADS, 256), 256, 0, stream>>>(ei, E, ET, as1, ad1, m1);
  k_edge_acc1<<<cdiv(ET * HEADS, 256), 256, 0, stream>>>(ei, E, ET, as1, ad1, m1, h1, dn1, o1);
  k_final1<<<cdiv((long long)N * H1DIM, 256), 256, 0, stream>>>(o1, dn1, b1, N);

  // ---- layer 2 ----
  k_proj2<<<cdiv(N, 256), 256, 0, stream>>>(o1, W2, asrc2, adst2, h2, as2, ad2, m2, dn2, o2, N);
  k_edge_max2<<<cdiv(ET, 256), 256, 0, stream>>>(ei, E, ET, as2, ad2, m2);
  k_edge_acc2<<<cdiv(ET, 256), 256, 0, stream>>>(ei, E, ET, as2, ad2, m2, h2, dn2, o2);

  // ---- pool + log_softmax ----
  k_pool_init<<<1, 256, 0, stream>>>(pooled, cnt, G);
  k_pool<<<cdiv(N, 256), 256, 0, stream>>>(o2, dn2, b2, batch, pooled, cnt, N);
  k_logsm<<<1, 64, 0, stream>>>(pooled, cnt, out, G);
}
